// GNNSTANHybrid_4621384810733
// MI455X (gfx1250) — matrix operand order repaired and checked by reading, not compiled, not run on hardware
//
#include <hip/hip_runtime.h>
#include <hip/hip_bf16.h>

#define EPSV 1e-5f
#define NB 64
#define RN 200
#define TT 256
#define DM 256
#define NEDGE (NB*RN*32)

typedef __attribute__((ext_vector_type(16))) _Float16 v16h;
typedef __attribute__((ext_vector_type(8)))  _Float16 v8h;
typedef __attribute__((ext_vector_type(4)))  _Float16 v4h;
typedef __attribute__((ext_vector_type(8)))  float    v8f;
typedef __attribute__((ext_vector_type(4)))  float    v4f;

#define AST 40   // LDS row stride (halves) for A tile: 80B -> 16B aligned rows
#define BST 40   // LDS row stride (halves) for Bt tile

// ------------------------------------------------------------------
// LDS-staged WMMA GEMM: C = act(alpha * A@B (+ bias)), f32 in/out, f16 MACs.
// Block tile 64x128, 8 waves (2x4), each wave 32x32 = 2x2 WMMA tiles.
// Two-level batch: z -> (bo = z/batchH, hi = z%batchH), offsets via strides.
// ------------------------------------------------------------------
__global__ void k_wmma_gemm(const float* __restrict__ A, const float* __restrict__ Bm,
                            const float* __restrict__ bias, float* __restrict__ C,
                            int M, int N, int K, int lda, int ldb, int ldc, int transB,
                            int batchH, long sA1, long sA2, long sB1, long sB2,
                            long sC1, long sC2, float alpha, int act)
{
  __shared__ _Float16 As[64 * AST];
  __shared__ _Float16 Bt[128 * BST];

  int z = blockIdx.z;
  int bo = z / batchH, hi = z - bo * batchH;
  A  += (long)bo * sA1 + (long)hi * sA2;
  Bm += (long)bo * sB1 + (long)hi * sB2;
  C  += (long)bo * sC1 + (long)hi * sC2;

  int tilesN = (N + 127) >> 7;
  int bmi = blockIdx.x / tilesN, bni = blockIdx.x - bmi * tilesN;
  int m0 = bmi * 64, n0 = bni * 128;

  int tid = threadIdx.x;
  int wave = tid >> 5, lane = tid & 31;
  int half = lane >> 4, l16 = lane & 15;
  int wm = wave >> 2, wn = wave & 3;

  v8f zero = {0.f,0.f,0.f,0.f,0.f,0.f,0.f,0.f};
  v8f acc[2][2];
  acc[0][0] = zero; acc[0][1] = zero; acc[1][0] = zero; acc[1][1] = zero;

  for (int k0 = 0; k0 < K; k0 += 32) {
    bool full = (m0 + 64 <= M) && (n0 + 128 <= N) && (k0 + 32 <= K);
    if (full) {
      int kn = k0 + 32;                 // prefetch next K chunk (global_prefetch_b8)
      bool pf = kn + 32 <= K;
      // ---- A tile 64x32 via float4 (all lda used are multiples of 4 floats)
#pragma unroll
      for (int j = 0; j < 2; ++j) {
        int i = tid + j * 256;
        int r = i >> 3, kq = i & 7;
        const float* ap = A + (long)(m0 + r) * lda + k0 + kq * 4;
        v4f v = *(const v4f*)ap;
        if (pf) __builtin_prefetch(ap + 32, 0, 1);
        v4h h4; h4[0] = (_Float16)v[0]; h4[1] = (_Float16)v[1];
        h4[2] = (_Float16)v[2]; h4[3] = (_Float16)v[3];
        *(v4h*)(As + r * AST + kq * 4) = h4;
      }
      // ---- B tile 32x128 -> Bt[n][k]
      if (transB) {
#pragma unroll
        for (int j = 0; j < 4; ++j) {
          int i = tid + j * 256;
          int r = i >> 3, kq = i & 7;     // r = column n, contiguous k reads
          const float* bp = Bm + (long)(n0 + r) * ldb + k0 + kq * 4;
          v4f v = *(const v4f*)bp;
          if (pf) __builtin_prefetch(bp + 32, 0, 1);
          v4h h4; h4[0] = (_Float16)v[0]; h4[1] = (_Float16)v[1];
          h4[2] = (_Float16)v[2]; h4[3] = (_Float16)v[3];
          *(v4h*)(Bt + r * BST + kq * 4) = h4;
        }
      } else {
#pragma unroll
        for (int j = 0; j < 4; ++j) {
          int i = tid + j * 256;
          int kr = i >> 5, nq = i & 31;   // coalesced along n, transpose on store
          const float* bp = Bm + (long)(k0 + kr) * ldb + n0 + nq * 4;
          v4f v = *(const v4f*)bp;
          if (pf) __builtin_prefetch(bp + (long)32 * ldb, 0, 1);
#pragma unroll
          for (int l = 0; l < 4; ++l) Bt[(nq * 4 + l) * BST + kr] = (_Float16)v[l];
        }
      }
    } else {
      // ---- edge path: clamped loads, select-to-zero
      for (int i = tid; i < 64 * 32; i += 256) {
        int r = i >> 5, kk = i & 31;
        int gm = m0 + r, gk = k0 + kk;
        float v = A[(long)min(gm, M - 1) * lda + min(gk, K - 1)];
        As[r * AST + kk] = (_Float16)((gm < M && gk < K) ? v : 0.f);
      }
      for (int i = tid; i < 128 * 32; i += 256) {
        int r = i >> 5, kk = i & 31;
        int gn = n0 + r, gk = k0 + kk;
        long off = transB ? ((long)min(gn, N - 1) * ldb + min(gk, K - 1))
                          : ((long)min(gk, K - 1) * ldb + min(gn, N - 1));
        float v = Bm[off];
        Bt[r * BST + kk] = (_Float16)((gn < N && gk < K) ? v : 0.f);
      }
    }
    __syncthreads();

    // ---- fragment loads from LDS (contiguous 16B chunks) + 4 WMMAs
    v16h af[2], bf[2];
#pragma unroll
    for (int mi = 0; mi < 2; ++mi) {
      const _Float16* ap = As + (wm * 32 + mi * 16 + l16) * AST + half * 8;
      v8h a0 = *(const v8h*)ap;
      v8h a1 = *(const v8h*)(ap + 16);
      af[mi] = __builtin_shufflevector(a0, a1, 0,1,2,3,4,5,6,7,8,9,10,11,12,13,14,15);
    }
#pragma unroll
    for (int ni = 0; ni < 2; ++ni) {
      const _Float16* bp = Bt + (wn * 32 + ni * 16 + l16) * BST + half * 8;
      v8h b0 = *(const v8h*)bp;
      v8h b1 = *(const v8h*)(bp + 16);
      bf[ni] = __builtin_shufflevector(b0, b1, 0,1,2,3,4,5,6,7,8,9,10,11,12,13,14,15);
    }
#pragma unroll
    for (int mi = 0; mi < 2; ++mi)
#pragma unroll
      for (int ni = 0; ni < 2; ++ni)
        acc[mi][ni] = __builtin_amdgcn_wmma_f32_16x16x32_f16(
            false, af[mi], false, bf[ni], (short)0, acc[mi][ni], false, false);
    __syncthreads();
  }

  // ---- epilogue
#pragma unroll
  for (int mi = 0; mi < 2; ++mi)
#pragma unroll
    for (int ni = 0; ni < 2; ++ni)
#pragma unroll
      for (int r = 0; r < 8; ++r) {
        int mm = m0 + wm * 32 + mi * 16 + r + half * 8;
        int nn = n0 + wn * 32 + ni * 16 + l16;
        if (mm < M && nn < N) {
          float v = acc[mi][ni][r] * alpha;
          if (bias) v += bias[nn];
          if (act == 1) v = fmaxf(v, 0.f);
          C[(long)mm * ldc + nn] = v;
        }
      }
}

static inline void gemm(hipStream_t st, const float* A, const float* B, const float* bias,
                        float* C, int M, int N, int K, int lda, int ldb, int ldc, int transB,
                        float alpha = 1.f, int act = 0,
                        int batch = 1, int batchH = 1,
                        long sA1 = 0, long sA2 = 0, long sB1 = 0, long sB2 = 0,
                        long sC1 = 0, long sC2 = 0)
{
  int tiles = ((M + 63) / 64) * ((N + 127) / 128);
  dim3 grid(tiles, 1, batch);
  k_wmma_gemm<<<grid, dim3(256), 0, st>>>(A, B, bias, C, M, N, K, lda, ldb, ldc, transB,
                                          batchH, sA1, sA2, sB1, sB2, sC1, sC2, alpha, act);
}

// ------------------------------------------------------------------ utils
__global__ void k_fill_f32(float* p, float v, long n) {
  long i = (long)blockIdx.x * 256 + threadIdx.x; if (i < n) p[i] = v;
}
__global__ void k_fill_u32(unsigned* p, unsigned v, long n) {
  long i = (long)blockIdx.x * 256 + threadIdx.x; if (i < n) p[i] = v;
}
__device__ inline unsigned fenc(float f) { unsigned u = __float_as_uint(f); return (u & 0x80000000u) ? ~u : (u | 0x80000000u); }
__device__ inline float fdec(unsigned u) { return __uint_as_float((u & 0x80000000u) ? (u ^ 0x80000000u) : ~u); }
__device__ inline float lrelu(float x) { return x > 0.f ? x : 0.2f * x; }
__device__ inline float sigm(float x) { return 1.f / (1.f + expf(-x)); }

// ------------------------------------------------------------------ GNN kernels
__global__ void k_init_edges(const int* __restrict__ ei, int* src2, int* dst2, int* mk) {
  int e = blockIdx.x * 256 + threadIdx.x; if (e >= NEDGE) return;
  src2[e] = ei[e]; dst2[e] = ei[NEDGE + e]; mk[e] = 1;
}

__global__ void k_gat_alpha(const float* __restrict__ h2, const float* __restrict__ a_s,
                            const float* __restrict__ a_d, float* asrc, float* adst, int n, int dd) {
  int i = blockIdx.x * 256 + threadIdx.x; if (i >= n * 4) return;
  int node = i >> 2, hd = i & 3;
  const float* hp = h2 + ((long)node * 4 + hd) * dd;
  float ss = 0.f, dv = 0.f;
  for (int c = 0; c < dd; ++c) { float v = hp[c]; ss += v * a_s[hd * dd + c]; dv += v * a_d[hd * dd + c]; }
  asrc[i] = ss; adst[i] = dv;
}

__global__ void k_edge_max(const int* src2, const int* dst2, const int* mk,
                           const float* asrc, const float* adst, unsigned* segmax, int n) {
  int idx = blockIdx.x * 256 + threadIdx.x;
  if (idx < NEDGE) {
    if (!mk[idx]) return;
    int s = src2[idx], d = dst2[idx];
    for (int hd = 0; hd < 4; ++hd)
      atomicMax(&segmax[d * 4 + hd], fenc(lrelu(asrc[s * 4 + hd] + adst[d * 4 + hd])));
  } else {
    int i = idx - NEDGE; if (i >= n) return;
    for (int hd = 0; hd < 4; ++hd)
      atomicMax(&segmax[i * 4 + hd], fenc(lrelu(asrc[i * 4 + hd] + adst[i * 4 + hd])));
  }
}

__global__ void k_edge_exp(const int* src2, const int* dst2, const int* mk,
                           const float* asrc, const float* adst, const unsigned* segmax,
                           float* edge_e, float* selfe, float* segsum, int n) {
  int idx = blockIdx.x * 256 + threadIdx.x;
  if (idx < NEDGE) {
    if (!mk[idx]) return;
    int s = src2[idx], d = dst2[idx];
    for (int hd = 0; hd < 4; ++hd) {
      float al = lrelu(asrc[s * 4 + hd] + adst[d * 4 + hd]);
      float e = expf(al - fdec(segmax[d * 4 + hd]));
      edge_e[(long)idx * 4 + hd] = e;
      atomicAdd(&segsum[d * 4 + hd], e);
    }
  } else {
    int i = idx - NEDGE; if (i >= n) return;
    for (int hd = 0; hd < 4; ++hd) {
      float al = lrelu(asrc[i * 4 + hd] + adst[i * 4 + hd]);
      float e = expf(al - fdec(segmax[i * 4 + hd]));
      selfe[i * 4 + hd] = e;
      atomicAdd(&segsum[i * 4 + hd], e);
    }
  }
}

// grid = NEDGE + n blocks, blockDim = dd
__global__ void k_gat_agg(const int* src2, const int* dst2, const int* mk,
                          const float* edge_e, const float* selfe, const float* segsum,
                          const float* __restrict__ h2, float* agg, int n, int dd) {
  long idx = blockIdx.x; int c = threadIdx.x;
  int s, d; const float* ep;
  if (idx < NEDGE) {
    if (!mk[idx]) return;
    s = src2[idx]; d = dst2[idx]; ep = &edge_e[idx * 4];
  } else {
    int i = (int)(idx - NEDGE); if (i >= n) return;
    s = i; d = i; ep = &selfe[i * 4];
  }
  float acc = 0.f;
  for (int hd = 0; hd < 4; ++hd) {
    float a = ep[hd] / (segsum[d * 4 + hd] + 1e-16f);
    acc += a * h2[((long)s * 4 + hd) * dd + c];
  }
  atomicAdd(&agg[(long)d * dd + c], acc);
}

__global__ void k_gat_final(const float* agg, const float* gb, const float* bng, const float* bnb,
                            const float* hres, float* hout, int n, int dd) {
  long i = (long)blockIdx.x * 256 + threadIdx.x; if (i >= (long)n * dd) return;
  int c = (int)(i % dd);
  float sc = rsqrtf(1.f + EPSV);
  float x = agg[i] * 0.25f + gb[c];
  x = x * (bng[c] * sc) + bnb[c] + hres[i];
  hout[i] = fmaxf(x, 0.f);
}

// one block (256 thr) per graph
__global__ void k_topk(const float* __restrict__ h, const float* __restrict__ p,
                       float* xnew, int* newid, int n_g, int k, int dim) {
  __shared__ float sc[256];
  __shared__ float red[256];
  int b = blockIdx.x, tid = threadIdx.x;
  float v = (tid < dim) ? p[tid] * p[tid] : 0.f;
  red[tid] = v; __syncthreads();
  for (int s = 128; s > 0; s >>= 1) { if (tid < s) red[tid] += red[tid + s]; __syncthreads(); }
  float pn = sqrtf(red[0]);
  if (tid < n_g) {
    const float* hp = h + ((long)b * n_g + tid) * dim;
    float d = 0.f;
    for (int c = 0; c < dim; ++c) d += hp[c] * p[c];
    sc[tid] = tanhf(d / pn);
  }
  __syncthreads();
  if (tid < n_g) {
    float si = sc[tid]; int cnt = 0;
    for (int j = 0; j < n_g; ++j) { float sj = sc[j]; cnt += (sj > si) || (sj == si && j < tid); }
    int gid = b * n_g + tid;
    if (cnt < k) {
      newid[gid] = b * k + cnt;
      const float* hp = h + (long)gid * dim;
      float* xp = xnew + ((long)b * k + cnt) * dim;
      for (int c = 0; c < dim; ++c) xp[c] = hp[c] * si;
    } else newid[gid] = -1;
  }
}

__global__ void k_remap(int* src2, int* dst2, int* mk, const int* newid) {
  int e = blockIdx.x * 256 + threadIdx.x; if (e >= NEDGE) return;
  int ns = newid[src2[e]], nd = newid[dst2[e]];
  mk[e] = mk[e] & (ns >= 0) & (nd >= 0);
  src2[e] = ns >= 0 ? ns : 0;
  dst2[e] = nd >= 0 ? nd : 0;
}

__global__ void k_gcn_deg(const int* src2, const int* dst2, const int* mk, float* deg, int n) {
  int idx = blockIdx.x * 256 + threadIdx.x;
  if (idx < NEDGE) { if (mk[idx]) atomicAdd(&deg[dst2[idx]], 1.f); }
  else { int i = idx - NEDGE; if (i < n) atomicAdd(&deg[i], 1.f); }
}

// grid NEDGE+n blocks, blockDim 32
__global__ void k_gcn_agg(const int* src2, const int* dst2, const int* mk,
                          const float* deg, const float* __restrict__ hW, float* agg, int n) {
  long idx = blockIdx.x; int c = threadIdx.x;
  int s, d;
  if (idx < NEDGE) { if (!mk[idx]) return; s = src2[idx]; d = dst2[idx]; }
  else { int i = (int)(idx - NEDGE); if (i >= n) return; s = i; d = i; }
  float ds = deg[s], dd = deg[d];
  float coef = (ds > 0.f ? rsqrtf(ds) : 0.f) * (dd > 0.f ? rsqrtf(dd) : 0.f);
  atomicAdd(&agg[(long)d * 32 + c], coef * hW[(long)s * 32 + c]);
}

__global__ void k_gcn_final(const float* agg, const float* gb, const float* gng, const float* gnb,
                            float* hout, int n) {
  long i = (long)blockIdx.x * 256 + threadIdx.x; if (i >= (long)n * 32) return;
  int c = (int)(i & 31);
  float sc = rsqrtf(1.f + EPSV);
  float x = (agg[i] + gb[c]) * (gng[c] * sc) + gnb[c];
  hout[i] = fmaxf(x, 0.f);
}

__global__ void k_meanpool(const float* h, float* out, int n_g) {
  int i = blockIdx.x * 256 + threadIdx.x; if (i >= NB * 32) return;
  int b = i >> 5, c = i & 31;
  float s = 0.f;
  for (int j = 0; j < n_g; ++j) s += h[((long)b * n_g + j) * 32 + c];
  out[i] = s / (float)n_g;
}

// ------------------------------------------------------------------ STAN kernels
__global__ void k_transpose_ts(const float* __restrict__ ts, float* out) {
  long i = (long)blockIdx.x * 256 + threadIdx.x;
  if (i >= (long)NB * TT * RN) return;
  int r = (int)(i % RN); long tmp = i / RN;
  int t = (int)(tmp % TT); int b = (int)(tmp / TT);
  out[i] = ts[((long)b * RN + r) * TT + t];
}

// one block per batch element; h/c/g live in LDS; 256 sequential steps
__global__ void k_lstm(const float* __restrict__ xg, const float* __restrict__ Whh,
                       const float* __restrict__ bih, const float* __restrict__ bhh,
                       float* __restrict__ hs, int rev, int ldo, int off) {
  __shared__ float hsh[128], csh[128], gsh[512];
  int b = blockIdx.x, tid = threadIdx.x;
  if (tid < 128) { hsh[tid] = 0.f; csh[tid] = 0.f; }
  __syncthreads();
  for (int step = 0; step < TT; ++step) {
    int t = rev ? (TT - 1 - step) : step;
    const float* xgp = xg + ((long)b * TT + t) * 512;
    for (int j = tid; j < 512; j += 256) {
      float a = xgp[j] + bih[j] + bhh[j];
      const float* w = Whh + (long)j * 128;
      for (int kk = 0; kk < 128; ++kk) a += w[kk] * hsh[kk];
      gsh[j] = a;
    }
    __syncthreads();
    if (tid < 128) {
      float ig = sigm(gsh[tid]);
      float fg = sigm(gsh[128 + tid]);
      float gg = tanhf(gsh[256 + tid]);
      float og = sigm(gsh[384 + tid]);
      float c = fg * csh[tid] + ig * gg;
      float hh = og * tanhf(c);
      csh[tid] = c; hsh[tid] = hh;
      hs[((long)b * TT + t) * ldo + off + tid] = hh;
    }
    __syncthreads();
  }
}

__global__ void k_softmax_row(float* x) {  // rows of length 256, in place
  __shared__ float red[256];
  long base = (long)blockIdx.x * 256;
  int tid = threadIdx.x;
  float v = x[base + tid];
  red[tid] = v; __syncthreads();
  for (int s = 128; s > 0; s >>= 1) { if (tid < s) red[tid] = fmaxf(red[tid], red[tid + s]); __syncthreads(); }
  float mx = red[0]; __syncthreads();
  float e = expf(v - mx);
  red[tid] = e; __syncthreads();
  for (int s = 128; s > 0; s >>= 1) { if (tid < s) red[tid] += red[tid + s]; __syncthreads(); }
  x[base + tid] = e / red[0];
}

__global__ void k_ln(const float* __restrict__ t0, const float* __restrict__ resid,
                     const float* __restrict__ g, const float* __restrict__ b,
                     float* __restrict__ o) {
  __shared__ float red[256];
  long base = (long)blockIdx.x * 256;
  int tid = threadIdx.x;
  float x = t0[base + tid] + resid[base + tid];
  red[tid] = x; __syncthreads();
  for (int s = 128; s > 0; s >>= 1) { if (tid < s) red[tid] += red[tid + s]; __syncthreads(); }
  float mean = red[0] * (1.f / 256.f); __syncthreads();
  float d = x - mean;
  red[tid] = d * d; __syncthreads();
  for (int s = 128; s > 0; s >>= 1) { if (tid < s) red[tid] += red[tid + s]; __syncthreads(); }
  float var = red[0] * (1.f / 256.f);
  o[base + tid] = d * rsqrtf(var + EPSV) * g[tid] + b[tid];
}

__global__ void k_repack_conv(const float* __restrict__ cw, float* W2) {
  int i = blockIdx.x * 256 + threadIdx.x; if (i >= 768 * 128) return;
  int o = i & 127, kidx = i >> 7;
  int kw = kidx / 256, ii = kidx - kw * 256;
  W2[i] = cw[((long)o * 256 + ii) * 3 + kw];   // conv_W (O=128, I=256, Kw=3)
}

__global__ void k_im2col(const float* __restrict__ att, float* X2) {
  long i = (long)blockIdx.x * 256 + threadIdx.x;
  if (i >= (long)NB * TT * 768) return;
  int kidx = (int)(i % 768); long row = i / 768;
  int t = (int)(row % TT); int b = (int)(row / TT);
  int kw = kidx / 256, ii = kidx - kw * 256;
  int tt = t + kw - 1;
  X2[i] = (tt >= 0 && tt < TT) ? att[((long)b * TT + tt) * DM + ii] : 0.f;
}

__global__ void k_mean_bn_relu(const float* __restrict__ conv, const float* g, const float* b,
                               float* out) {
  int i = blockIdx.x * 256 + threadIdx.x; if (i >= NB * 128) return;
  int bb = i >> 7, o = i & 127;
  float sc = g[o] * rsqrtf(1.f + EPSV), be = b[o];
  float s = 0.f;
  for (int t = 0; t < TT; ++t) {
    float v = conv[((long)bb * TT + t) * 128 + o] * sc + be;
    s += fmaxf(v, 0.f);
  }
  out[i] = s * (1.f / (float)TT);
}

// ------------------------------------------------------------------ fusion kernels
__global__ void k_concat4(const float* g, const float* s, const float* gc, const float* sc,
                          float* f4) {
  int i = blockIdx.x * 256 + threadIdx.x; if (i >= NB * 512) return;
  int b = i >> 9, j = i & 511;
  float v;
  if (j < 128)      v = g[b * 128 + j];
  else if (j < 256) v = s[b * 128 + j - 128];
  else if (j < 384) v = gc[b * 128 + j - 256];
  else              v = sc[b * 128 + j - 384];
  f4[i] = v;
}

__global__ void k_bn_elem(float* x, const float* g, const float* b, int C, long n) {
  long i = (long)blockIdx.x * 256 + threadIdx.x; if (i >= n) return;
  int c = (int)(i % C);
  x[i] = x[i] * (g[c] * rsqrtf(1.f + EPSV)) + b[c];
}

__global__ void k_softmax2(const float* logits, float* out) {
  int b = blockIdx.x * 256 + threadIdx.x; if (b >= NB) return;
  float a = logits[b * 2], c = logits[b * 2 + 1];
  float m = fmaxf(a, c);
  float ea = expf(a - m), ec = expf(c - m);
  float s = ea + ec;
  out[b * 2] = ea / s; out[b * 2 + 1] = ec / s;
}

// ------------------------------------------------------------------ driver
extern "C" void kernel_launch(void* const* d_in, const int* in_sizes, int n_in,
                              void* d_out, int out_size, void* d_ws, size_t ws_size,
                              hipStream_t st) {
  (void)in_sizes; (void)n_in; (void)out_size; (void)ws_size;
  const float* fc = (const float*)d_in[0];
  const float* ts = (const float*)d_in[1];
  const int* ei = (const int*)d_in[2];
  auto P = [&](int i) { return (const float*)d_in[4 + i]; };
  float* out = (float*)d_out;
  float* Wf = (float*)d_ws;

  // ---------------- GNN arena ----------------
  float* nodeA = Wf;
  float* nodeB = nodeA + 2560000;
  float* hres  = nodeB + 2560000;
  float* h2    = hres + 1638400;
  float* agg   = h2 + 6553600;
  float* asrc  = agg + 1638400;
  float* adst  = asrc + 51200;
  float* segmaxf = adst + 51200;
  float* segsum  = segmaxf + 51200;
  float* selfe   = segsum + 51200;
  float* edge_e  = selfe + 51200;
  float* deg     = edge_e + 1638400;
  int* newid = (int*)(deg + 12800);
  int* src2  = newid + 12800;
  int* dst2  = src2 + NEDGE;
  int* maskb = dst2 + NEDGE;
  unsigned* segmax = (unsigned*)segmaxf;

  unsigned encneg; { union { float f; unsigned u; } cv; cv.f = -1e9f; encneg = ~cv.u; }

  k_init_edges<<<(NEDGE + 255) / 256, 256, 0, st>>>(ei, src2, dst2, maskb);

  const int DI[3] = {200, 128, 64};
  const int DO[3] = {128, 64, 32};
  const int KP[2] = {160, 96};
  int n = NB * RN, ng = RN;
  const float* hcur = fc;

  for (int i = 0; i < 3; ++i) {
    int di = DI[i], dd = DO[i];
    gemm(st, hcur, P(8 * i + 6), P(8 * i + 7), hres, n, dd, di, di, dd, dd, 0);
    gemm(st, hcur, P(8 * i + 0), nullptr, h2, n, 4 * dd, di, di, 4 * dd, 4 * dd, 0);
    k_gat_alpha<<<(n * 4 + 255) / 256, 256, 0, st>>>(h2, P(8 * i + 1), P(8 * i + 2), asrc, adst, n, dd);
    k_fill_u32<<<(n * 4 + 255) / 256, 256, 0, st>>>(segmax, encneg, n * 4);
    k_fill_f32<<<(n * 4 + 255) / 256, 256, 0, st>>>(segsum, 0.f, n * 4);
    k_fill_f32<<<((long)n * dd + 255) / 256, 256, 0, st>>>(agg, 0.f, (long)n * dd);
    int tot = NEDGE + n;
    k_edge_max<<<(tot + 255) / 256, 256, 0, st>>>(src2, dst2, maskb, asrc, adst, segmax, n);
    k_edge_exp<<<(tot + 255) / 256, 256, 0, st>>>(src2, dst2, maskb, asrc, adst, segmax, edge_e, selfe, segsum, n);
    k_gat_agg<<<tot, dd, 0, st>>>(src2, dst2, maskb, edge_e, selfe, segsum, h2, agg, n, dd);
    k_gat_final<<<((long)n * dd + 255) / 256, 256, 0, st>>>(agg, P(8 * i + 3), P(8 * i + 4), P(8 * i + 5), hres, nodeA, n, dd);
    hcur = nodeA;
    if (i < 2) {
      int kk = KP[i];
      k_topk<<<NB, 256, 0, st>>>(hcur, P(24 + i), nodeB, newid, ng, kk, dd);
      k_remap<<<(NEDGE + 255) / 256, 256, 0, st>>>(src2, dst2, maskb, newid);
      ng = kk; n = NB * kk; hcur = nodeB;
    }
  }
  // GCN + pool
  k_fill_f32<<<(n + 255) / 256, 256, 0, st>>>(deg, 0.f, n);
  k_gcn_deg<<<(NEDGE + n + 255) / 256, 256, 0, st>>>(src2, dst2, maskb, deg, n);
  gemm(st, hcur, P(26), nullptr, h2, n, 32, 32, 32, 32, 32, 0);
  k_fill_f32<<<((long)n * 32 + 255) / 256, 256, 0, st>>>(agg, 0.f, (long)n * 32);
  k_gcn_agg<<<NEDGE + n, 32, 0, st>>>(src2, dst2, maskb, deg, h2, agg, n);
  k_gcn_final<<<((long)n * 32 + 255) / 256, 256, 0, st>>>(agg, P(27), P(28), P(29), nodeB, n);
  k_meanpool<<<(NB * 32 + 255) / 256, 256, 0, st>>>(nodeB, out + 256, ng);

  // ---------------- STAN arena (reuses GNN region; stream-ordered) ----------------
  float* hswap = Wf;
  float* xg    = hswap + 3276800;
  float* seqA  = xg + 8388608;
  float* seqB  = seqA + 4194304;
  float* Qb    = seqB + 4194304;
  float* Kb    = Qb + 4194304;
  float* Vb    = Kb + 4194304;
  float* scoresb = Vb + 4194304;
  float* ctxb  = scoresb + 4194304;
  float* attb  = ctxb + 4194304;
  float* X2 = Qb;         // reuse Q/K/V (12,582,912 floats exactly)
  float* W2 = scoresb;    // reuse
  float* convout = ctxb;  // reuse

  k_transpose_ts<<<(unsigned)(((long)NB * TT * RN + 255) / 256), 256, 0, st>>>(ts, hswap);

  // bi-LSTM layer 0 (input dim 200)
  for (int d = 0; d < 2; ++d) {
    int base = 30 + d * 4;
    gemm(st, hswap, P(base + 0), nullptr, xg, NB * TT, 512, RN, RN, RN, 512, 1);
    k_lstm<<<NB, 256, 0, st>>>(xg, P(base + 1), P(base + 2), P(base + 3), seqA, d, 256, d * 128);
  }
  // bi-LSTM layer 1 (input dim 256)
  for (int d = 0; d < 2; ++d) {
    int base = 30 + 8 + d * 4;
    gemm(st, seqA, P(base + 0), nullptr, xg, NB * TT, 512, 256, 256, 256, 512, 1);
    k_lstm<<<NB, 256, 0, st>>>(xg, P(base + 1), P(base + 2), P(base + 3), seqB, d, 256, d * 128);
  }

  // multi-head self attention
  gemm(st, seqB, P(46), nullptr, Qb, NB * TT, 256, 256, 256, 256, 256, 0);
  gemm(st, seqB, P(47), nullptr, Kb, NB * TT, 256, 256, 256, 256, 256, 0);
  gemm(st, seqB, P(48), nullptr, Vb, NB * TT, 256, 256, 256, 256, 256, 0);
  const float rs = 0.17677669529663687f;  // 1/sqrt(32)
  for (int cb = 0; cb < 8; ++cb) {
    const float* Qc = Qb + (long)cb * 8 * TT * DM;
    const float* Kc = Kb + (long)cb * 8 * TT * DM;
    const float* Vc = Vb + (long)cb * 8 * TT * DM;
    // scores = Q @ K^T * rs   (8 batches x 8 heads)
    gemm(st, Qc, Kc, nullptr, scoresb, 256, 256, 32, 256, 256, 256, 1, rs, 0,
         64, 8, (long)TT * DM, 32, (long)TT * DM, 32, (long)8 * TT * 256, (long)TT * 256);
    k_softmax_row<<<64 * 256, 256, 0, st>>>(scoresb);
    // ctx = softmax(scores) @ V  -> written straight into (b,t,h*32+d) layout
    gemm(st, scoresb, Vc, nullptr, ctxb + (long)cb * 8 * TT * DM, 256, 32, 256, 256, 256, 256, 0, 1.f, 0,
         64, 8, (long)8 * TT * 256, (long)TT * 256, (long)TT * DM, 32, (long)TT * DM, 32);
  }
  // out proj + residual + layernorm
  gemm(st, ctxb, P(49), P(50), xg, NB * TT, 256, 256, 256, 256, 256, 0);
  k_ln<<<NB * TT, 256, 0, st>>>(xg, seqB, P(51), P(52), attb);

  // temporal conv (im2col + GEMM), then bn+relu+mean over time -> s_emb
  k_repack_conv<<<(768 * 128 + 255) / 256, 256, 0, st>>>(P(53), W2);
  k_im2col<<<(unsigned)(((long)NB * TT * 768 + 255) / 256), 256, 0, st>>>(attb, X2);
  gemm(st, X2, W2, P(54), convout, NB * TT, 128, 768, 768, 128, 128, 0);
  k_mean_bn_relu<<<(NB * 128 + 255) / 256, 256, 0, st>>>(convout, P(55), P(56), out + 2304);

  // ---------------- fusion + classifier ----------------
  float* gv  = Wf;           // tiny buffers reuse the start of the arena
  float* sv  = gv + 8192;
  float* gcv = sv + 8192;
  float* scv = gcv + 8192;
  float* f4  = scv + 8192;
  float* fbuf = f4 + 32768;
  float* hcls = fbuf + 16384;
  float* tv   = hcls + 4096;

  gemm(st, out + 256, P(57), P(58), gv, 64, 128, 32, 32, 128, 128, 0);
  gemm(st, out + 2304, P(59), P(60), sv, 64, 128, 128, 128, 128, 128, 0);
  // len-1 MHA degenerates to out_proj(v_proj(x))
  gemm(st, sv, P(61) + 256 * 128, P(62) + 256, tv, 64, 128, 128, 128, 128, 128, 1);
  gemm(st, tv, P(63), P(64), gcv, 64, 128, 128, 128, 128, 128, 1);
  gemm(st, gv, P(65) + 256 * 128, P(66) + 256, tv, 64, 128, 128, 128, 128, 128, 1);
  gemm(st, tv, P(67), P(68), scv, 64, 128, 128, 128, 128, 128, 1);
  k_concat4<<<(NB * 512 + 255) / 256, 256, 0, st>>>(gv, sv, gcv, scv, f4);
  gemm(st, f4, P(69), P(70), fbuf, 64, 256, 512, 512, 256, 256, 0, 1.f, 1);
  k_bn_elem<<<(NB * 256 + 255) / 256, 256, 0, st>>>(fbuf, P(71), P(72), 256, NB * 256);
  gemm(st, fbuf, P(73), P(74), out + 10496, 64, 128, 256, 256, 128, 128, 0, 1.f, 1);  // fused
  gemm(st, out + 10496, P(75), P(76), hcls, 64, 64, 128, 128, 64, 64, 0, 1.f, 1);
  k_bn_elem<<<(NB * 64 + 255) / 256, 256, 0, st>>>(hcls, P(77), P(78), 64, NB * 64);
  gemm(st, hcls, P(79), P(80), out, 64, 2, 64, 64, 2, 2, 0);  // logits
  k_softmax2<<<1, 256, 0, st>>>(out, out + 128);
}
